// GRULayer_84670985273420
// MI455X (gfx1250) — matrix operand order repaired and checked by reading, not compiled, not run on hardware
//
#include <hip/hip_runtime.h>

// GRU: B=16, T=1024, E=512, H=1024
#define B_  16
#define T_  1024
#define E_  512
#define H_  1024
#define BT_ 16384   // B*T

typedef _Float16 half_t;
typedef __attribute__((ext_vector_type(8)))  _Float16 v8h;
typedef __attribute__((ext_vector_type(16))) _Float16 v16h;
typedef __attribute__((ext_vector_type(8)))  float    v8f;

// ---------------- workspace layout (bytes) ----------------
#define OFF_CNT   0u
#define OFF_HF32  256u                                  // float h[2][16][1024]
#define OFF_HF16  (OFF_HF32 + 2u*16u*1024u*4u)          // half  h[2][16][1024]
#define OFF_ZERO_END (OFF_HF16 + 2u*16u*1024u*2u)       // memset range end
#define OFF_XH    ((size_t)OFF_ZERO_END)                // half x[BT][E]
#define OFF_WXT   (OFF_XH  + (size_t)BT_*E_*2)          // half WxT[3][H][E]
#define OFF_WHT   (OFF_WXT + (size_t)3*H_*E_*2)         // half WhT[3][H][H]
#define OFF_PRE   (OFF_WHT + (size_t)3*H_*H_*2)         // half pre[3][BT][H]

// ---------------- helpers ----------------
__device__ __forceinline__ v16h make_a16(const half_t* row, int k0, int hi) {
    // A-matrix 16x32 f16, contiguous k order (B matches): lane<16 -> K {k0..k0+15}
    //                                     lane>=16 -> K {k0+16..k0+31}
    v8h a0 = *(const v8h*)(row + k0 + hi * 16);
    v8h a1 = *(const v8h*)(row + k0 + hi * 16 + 8);
    return __builtin_shufflevector(a0, a1, 0,1,2,3,4,5,6,7,8,9,10,11,12,13,14,15);
}

// ---------------- kernel 1: fp32 -> f16 convert of x ----------------
__global__ void gru_cvt_x(const float* __restrict__ x, half_t* __restrict__ xh, int n) {
    int idx = blockIdx.x * blockDim.x + threadIdx.x;
    if (idx < n) xh[idx] = (half_t)x[idx];
}

// ---------------- kernel 2: transpose weight fp32 [K][N] -> f16 [N][K] ----------------
__global__ void gru_trans_w(const float* __restrict__ W, half_t* __restrict__ WT, int K, int N) {
    int idx = blockIdx.x * blockDim.x + threadIdx.x;  // idx = n*K + k
    if (idx < N * K) {
        int n = idx / K, k = idx - n * K;
        WT[idx] = (half_t)W[(size_t)k * N + n];
    }
}

// ---------------- kernel 3: input projections via WMMA ----------------
// pre[g][bt][n] = (x @ Wx_g)[bt][n] + bias_g[n], f16
__global__ void __launch_bounds__(256) gru_in_gemm(
    const half_t* __restrict__ xh, const half_t* __restrict__ WxT,
    const float* __restrict__ br, const float* __restrict__ bz, const float* __restrict__ bb,
    half_t* __restrict__ pre) {
    const int lane = threadIdx.x & 31, w = threadIdx.x >> 5;
    const int row0 = blockIdx.x << 4;                    // 1024 row tiles
    const int g    = blockIdx.z;                         // gate
    const int col0 = (blockIdx.y << 9) + (w << 6);       // wave -> 16x64 tile
    const int m = lane & 15, hi = lane >> 4;

    const v8f vzero = {0.f,0.f,0.f,0.f,0.f,0.f,0.f,0.f};
    v8f acc[4];
    #pragma unroll
    for (int j = 0; j < 4; ++j) acc[j] = vzero;

    const half_t* arow  = xh + (size_t)(row0 + m) * E_;
    const half_t* wbase = WxT + (size_t)(g * H_ + col0) * E_;

    for (int k0 = 0; k0 < E_; k0 += 32) {
        v16h a = make_a16(arow, k0, hi);
        #pragma unroll
        for (int j = 0; j < 4; ++j) {
            // B-matrix 32x16: lane holds column n=lane&15; K contiguous (hi*16)
            v16h bm = *(const v16h*)(wbase + (size_t)(j * 16 + m) * E_ + k0 + hi * 16);
            acc[j] = __builtin_amdgcn_wmma_f32_16x16x32_f16(
                false, a, false, bm, (short)0, acc[j], false, false);
        }
    }
    const float* bias = (g == 0) ? br : (g == 1) ? bz : bb;
    #pragma unroll
    for (int j = 0; j < 4; ++j) {
        int col = col0 + j * 16 + m;
        float bv = bias[col];
        #pragma unroll
        for (int i = 0; i < 8; ++i) {
            int r = row0 + i + hi * 8;                   // C layout: M = i + 8*hi
            pre[((size_t)g * BT_ + r) * H_ + col] = (half_t)(acc[j][i] + bv);
        }
    }
}

// ---------------- kernel 4: persistent recurrent scan ----------------
// 64 blocks (one per 16 hidden columns) x 96 threads (3 waves = 3 gates).
__global__ void __launch_bounds__(96) gru_recurrent(
    const half_t* __restrict__ WhT, const half_t* __restrict__ pre,
    float* __restrict__ out, float* __restrict__ hf32, half_t* __restrict__ hf16,
    unsigned* __restrict__ counter) {
    __shared__ __align__(32) half_t wlds[3 * 16 * H_];   // 96 KB: this block's weight cols
    __shared__ float c3[3][16][16];

    const int tid = threadIdx.x, lane = tid & 31, w = tid >> 5;  // w = gate 0..2
    const int col0 = blockIdx.x << 4;
    const int m = lane & 15, hi = lane >> 4;

    // Preload weight columns for all 3 gates into LDS: wlds[g][n_local][k]
    for (int j = tid * 8; j < 3 * 16 * H_; j += 96 * 8) {
        int g = j >> 14, rem = j & 16383;
        int nl = rem >> 10, k = rem & 1023;
        *(v8h*)&wlds[j] = *(const v8h*)&WhT[(size_t)(g * H_ + col0 + nl) * H_ + k];
    }
    __syncthreads();

    const v8f vzero = {0.f,0.f,0.f,0.f,0.f,0.f,0.f,0.f};
    const half_t* wrow = &wlds[(w * 16 + m) * H_];       // this lane's B column

    for (int t = 0; t < T_; ++t) {
        const float* hprv = hf32 + (t & 1) * (16 * H_);

        // ---- hoisted loads (independent of h@W): in flight during the WMMA chain ----
        float xrv[3], xzv[3], xiv[3], hpv[3];
        #pragma unroll
        for (int it = 0; it < 3; ++it) {
            int idx = tid + it * 96;                     // wave-uniform guard
            if (idx < 256) {
                int mm = idx >> 4, nn = idx & 15, col = col0 + nn;
                size_t prow = (size_t)mm * T_ + t;
                xrv[it] = (float)pre[(prow) * H_ + col];
                xzv[it] = (float)pre[((size_t)BT_ + prow) * H_ + col];
                xiv[it] = (float)pre[((size_t)2 * BT_ + prow) * H_ + col];
                hpv[it] = hprv[mm * H_ + col];
            }
        }

        // ---- 16x16 gate tile: h @ Wh_g[:, col0:col0+16], dual accumulators ----
        const half_t* arow = hf16 + (size_t)(t & 1) * (16 * H_) + (size_t)m * H_;
        v8f acc0 = vzero, acc1 = vzero;
        #pragma unroll
        for (int k0 = 0; k0 < H_; k0 += 64) {
            v16h a0 = make_a16(arow, k0, hi);
            v16h b0 = *(const v16h*)(wrow + k0 + hi * 16);
            acc0 = __builtin_amdgcn_wmma_f32_16x16x32_f16(
                false, a0, false, b0, (short)0, acc0, false, false);
            v16h a1 = make_a16(arow, k0 + 32, hi);
            v16h b1 = *(const v16h*)(wrow + k0 + 32 + hi * 16);
            acc1 = __builtin_amdgcn_wmma_f32_16x16x32_f16(
                false, a1, false, b1, (short)0, acc1, false, false);
        }
        v8f acc = acc0 + acc1;
        #pragma unroll
        for (int i = 0; i < 8; ++i) c3[w][i + hi * 8][m] = acc[i];
        __syncthreads();

        float*  hdst   = hf32 + ((t + 1) & 1) * (16 * H_);
        half_t* hdst16 = hf16 + ((t + 1) & 1) * (16 * H_);
        #pragma unroll
        for (int it = 0; it < 3; ++it) {
            int idx = tid + it * 96;
            if (idx < 256) {
                int mm = idx >> 4, nn = idx & 15, col = col0 + nn;
                float rr = 1.f / (1.f + expf(-(xrv[it] + c3[0][mm][nn])));
                float zz = 1.f / (1.f + expf(-(xzv[it] + c3[1][mm][nn])));
                float cd = tanhf(rr * c3[2][mm][nn] + xiv[it]);
                float hn = zz * hpv[it] + (1.f - zz) * cd;
                hdst[mm * H_ + col]   = hn;
                hdst16[mm * H_ + col] = (half_t)hn;
                out[((size_t)mm * T_ + t) * H_ + col] = hn;
            }
        }
        // device-wide sync for step t
        __threadfence();
        __syncthreads();
        if (tid == 0)
            __hip_atomic_fetch_add(counter, 1u, __ATOMIC_RELAXED, __HIP_MEMORY_SCOPE_AGENT);
        unsigned target = (unsigned)(t + 1) * 64u;
        while (__hip_atomic_load(counter, __ATOMIC_ACQUIRE, __HIP_MEMORY_SCOPE_AGENT) < target) {
            __builtin_amdgcn_s_sleep(1);
        }
    }
}

// ---------------- launch ----------------
extern "C" void kernel_launch(void* const* d_in, const int* in_sizes, int n_in,
                              void* d_out, int out_size, void* d_ws, size_t ws_size,
                              hipStream_t stream) {
    const float* x   = (const float*)d_in[0];
    const float* Wxr = (const float*)d_in[1];
    const float* Whr = (const float*)d_in[2];
    const float* br  = (const float*)d_in[3];
    const float* Wxz = (const float*)d_in[4];
    const float* Whz = (const float*)d_in[5];
    const float* bz  = (const float*)d_in[6];
    const float* Wi  = (const float*)d_in[7];
    const float* Wh  = (const float*)d_in[8];
    const float* bb  = (const float*)d_in[9];
    float* out = (float*)d_out;

    char* ws = (char*)d_ws;
    unsigned* cnt  = (unsigned*)(ws + OFF_CNT);
    float*    hf32 = (float*)(ws + OFF_HF32);
    half_t*   hf16 = (half_t*)(ws + OFF_HF16);
    half_t*   xh   = (half_t*)(ws + OFF_XH);
    half_t*   WxT  = (half_t*)(ws + OFF_WXT);
    half_t*   WhT  = (half_t*)(ws + OFF_WHT);
    half_t*   pre  = (half_t*)(ws + OFF_PRE);

    // zero counter + both h buffers (graph-capturable)
    hipMemsetAsync(ws, 0, OFF_ZERO_END, stream);

    // fp32 -> f16 / transposed-weight prep
    gru_cvt_x<<<(BT_ * E_ + 255) / 256, 256, 0, stream>>>(x, xh, BT_ * E_);
    gru_trans_w<<<(H_ * E_ + 255) / 256, 256, 0, stream>>>(Wxr, WxT,                 E_, H_);
    gru_trans_w<<<(H_ * E_ + 255) / 256, 256, 0, stream>>>(Wxz, WxT + (size_t)H_*E_, E_, H_);
    gru_trans_w<<<(H_ * E_ + 255) / 256, 256, 0, stream>>>(Wi,  WxT + (size_t)2*H_*E_, E_, H_);
    gru_trans_w<<<(H_ * H_ + 255) / 256, 256, 0, stream>>>(Whr, WhT,                 H_, H_);
    gru_trans_w<<<(H_ * H_ + 255) / 256, 256, 0, stream>>>(Whz, WhT + (size_t)H_*H_, H_, H_);
    gru_trans_w<<<(H_ * H_ + 255) / 256, 256, 0, stream>>>(Wh,  WhT + (size_t)2*H_*H_, H_, H_);

    // batched input projections (WMMA)
    gru_in_gemm<<<dim3(BT_ / 16, H_ / 512, 3), 256, 0, stream>>>(xh, WxT, br, bz, bb, pre);

    // persistent recurrent kernel (WMMA + LDS weights + grid sync)
    gru_recurrent<<<64, 96, 0, stream>>>(WhT, pre, out, hf32, hf16, cnt);
}
